// SelfAttention_2929167696528
// MI455X (gfx1250) — matrix operand order repaired and checked by reading, not compiled, not run on hardware
//
#include <hip/hip_runtime.h>

typedef __attribute__((ext_vector_type(16))) _Float16 v16h;
typedef __attribute__((ext_vector_type(8)))  _Float16 v8h;
typedef __attribute__((ext_vector_type(8)))  float    v8f;
typedef __attribute__((ext_vector_type(4)))  float    v4f;

#define DEV static __device__ __forceinline__

constexpr int CB = 32;    // batch
constexpr int CC = 256;   // channels
constexpr int CN = 1024;  // H*W

DEV v8f vzero() {
  v8f z;
#pragma unroll
  for (int i = 0; i < 8; ++i) z[i] = 0.f;
  return z;
}

// Build a v16h operand from two 16-byte (b128) loads.
DEV v16h ld16(const _Float16* p0, const _Float16* p1) {
  union { v16h v; v8h h[2]; } u;
  u.h[0] = *(const v8h*)p0;
  u.h[1] = *(const v8h*)p1;
  return u.v;
}

DEV v8f wmma16(v16h a, v16h b, v8f c) {
  return __builtin_amdgcn_wmma_f32_16x16x32_f16(false, a, false, b, (short)0, c,
                                                false, false);
}

DEV float red16_max(float v) {
#pragma unroll
  for (int m = 1; m < 16; m <<= 1) v = fmaxf(v, __shfl_xor(v, m, 32));
  return v;
}
DEV float red16_sum(float v) {
#pragma unroll
  for (int m = 1; m < 16; m <<= 1) v += __shfl_xor(v, m, 32);
  return v;
}

// ---------------------------------------------------------------------------
// Kernel 0: convert all four weight matrices f32 -> f16 (L2-resident reuse).
// ---------------------------------------------------------------------------
__global__ void cvt_kernel(const float* __restrict__ wq, const float* __restrict__ wk,
                           const float* __restrict__ wv, const float* __restrict__ wp,
                           _Float16* __restrict__ dst) {
  int idx = blockIdx.x * 256 + threadIdx.x;           // 0 .. 65535
  dst[idx]               = (_Float16)wq[idx];
  dst[CC * CC + idx]     = (_Float16)wk[idx];
  dst[2 * CC * CC + idx] = (_Float16)wv[idx];
  dst[3 * CC * CC + idx] = (_Float16)wp[idx];
}

// ---------------------------------------------------------------------------
// Kernel 1: GroupNorm. One workgroup per (group, batch): 8 channels x 1024.
// Writes hT[b][n][c] f16 (channel-contiguous for WMMA B-operand loads).
// ---------------------------------------------------------------------------
__global__ void gn_kernel(const float* __restrict__ x, const float* __restrict__ gamma,
                          const float* __restrict__ beta, _Float16* __restrict__ hT) {
  int g = blockIdx.x, b = blockIdx.y;
  int t = threadIdx.x;
  int c0 = g * 8;
  const float* xb = x + (size_t)b * CC * CN;
  int n0 = t * 4;

  float vals[8][4];
  float s = 0.f, s2 = 0.f;
#pragma unroll
  for (int cc = 0; cc < 8; ++cc) {
    v4f v = *(const v4f*)(xb + (size_t)(c0 + cc) * CN + n0);
#pragma unroll
    for (int e = 0; e < 4; ++e) {
      float f = v[e];
      vals[cc][e] = f;
      s += f;
      s2 += f * f;
    }
  }
  __shared__ float rs[256], rs2[256];
  rs[t] = s; rs2[t] = s2;
  __syncthreads();
  for (int off = 128; off > 0; off >>= 1) {
    if (t < off) { rs[t] += rs[t + off]; rs2[t] += rs2[t + off]; }
    __syncthreads();
  }
  float mean = rs[0] * (1.f / 8192.f);
  float var  = rs2[0] * (1.f / 8192.f) - mean * mean;
  float rstd = rsqrtf(var + 1e-5f);

  float ga[8], be[8];
#pragma unroll
  for (int cc = 0; cc < 8; ++cc) {
    ga[cc] = gamma[c0 + cc] * rstd;
    be[cc] = beta[c0 + cc] - mean * ga[cc];
  }
  _Float16* hb = hT + (size_t)b * CN * CC;
#pragma unroll
  for (int e = 0; e < 4; ++e) {
    v8h o;
#pragma unroll
    for (int cc = 0; cc < 8; ++cc) o[cc] = (_Float16)(vals[cc][e] * ga[cc] + be[cc]);
    *(v8h*)(hb + (size_t)(n0 + e) * CC + c0) = o;
  }
}

// ---------------------------------------------------------------------------
// Kernel 2/4: WMMA matmul  out[c_out][n] = W @ inT + bias   (per batch).
// inT layout [N][C] f16. One workgroup: 256 rows x 64 cols; wave w owns rows
// [32w, 32w+32). MODE 0: f16 store transposed [n][c]; MODE 1: f16 [c][n];
// MODE 2: f32 [c][n] + residual.
// ---------------------------------------------------------------------------
template <int MODE>
__global__ void __launch_bounds__(256)
mm_kernel(const _Float16* __restrict__ w16, const float* __restrict__ bias,
          const _Float16* __restrict__ inT, _Float16* __restrict__ out16,
          float* __restrict__ outf, const float* __restrict__ resid, float mult) {
  int n0 = blockIdx.x * 64;
  int b  = blockIdx.y;
  int wid  = threadIdx.x >> 5;
  int lane = threadIdx.x & 31;
  int hl   = lane >> 4, lcol = lane & 15;
  int m0 = wid * 32;
  const _Float16* inb = inT + (size_t)b * CN * CC;

  v8f acc[2][4];
#pragma unroll
  for (int i = 0; i < 2; ++i)
#pragma unroll
    for (int j = 0; j < 4; ++j) acc[i][j] = vzero();

#pragma unroll
  for (int kc = 0; kc < CC; kc += 32) {
    v16h A[2], Bm[4];
#pragma unroll
    for (int i = 0; i < 2; ++i) {
      const _Float16* p = w16 + (size_t)(m0 + i * 16 + lcol) * CC + kc + 16 * hl;
      A[i] = ld16(p, p + 8);
    }
#pragma unroll
    for (int j = 0; j < 4; ++j) {
      const _Float16* p = inb + (size_t)(n0 + j * 16 + lcol) * CC + kc + 16 * hl;
      Bm[j] = ld16(p, p + 8);
    }
#pragma unroll
    for (int i = 0; i < 2; ++i)
#pragma unroll
      for (int j = 0; j < 4; ++j) acc[i][j] = wmma16(A[i], Bm[j], acc[i][j]);
  }

#pragma unroll
  for (int i = 0; i < 2; ++i) {
    int cb = m0 + i * 16 + 8 * hl;  // 8 consecutive output rows for this lane
#pragma unroll
    for (int j = 0; j < 4; ++j) {
      int n = n0 + j * 16 + lcol;
      if (MODE == 0) {
        v8h o;
#pragma unroll
        for (int r = 0; r < 8; ++r)
          o[r] = (_Float16)((acc[i][j][r] + bias[cb + r]) * mult);
        *(v8h*)(out16 + (size_t)b * CN * CC + (size_t)n * CC + cb) = o;
      } else if (MODE == 1) {
#pragma unroll
        for (int r = 0; r < 8; ++r)
          out16[(size_t)b * CC * CN + (size_t)(cb + r) * CN + n] =
              (_Float16)((acc[i][j][r] + bias[cb + r]) * mult);
      } else {
#pragma unroll
        for (int r = 0; r < 8; ++r) {
          size_t idx = (size_t)b * CC * CN + (size_t)(cb + r) * CN + n;
          outf[idx] = resid[idx] + (acc[i][j][r] + bias[cb + r]) * mult;
        }
      }
    }
  }
}

// ---------------------------------------------------------------------------
// Kernel 3: fused flash attention. One workgroup per (i-tile of 16, batch).
// 8 waves partition the j axis (each wave: 4 tiles of 32 columns), keep
// private online-softmax state + private O[256 x 16] f32 accumulators, then
// merge via LDS atomics. S = Q^T K and O += V P^T both via v_wmma f16.
// q is pre-scaled by C^-0.5 (folded into its projection epilogue).
// ---------------------------------------------------------------------------
__global__ void __launch_bounds__(256)
fa_kernel(const _Float16* __restrict__ qT, const _Float16* __restrict__ kT,
          const _Float16* __restrict__ vN, _Float16* __restrict__ ovT) {
  int i0 = blockIdx.x * 16;
  int b  = blockIdx.y;
  int t  = threadIdx.x;
  int wid = t >> 5, lane = t & 31, hl = lane >> 4, lcol = lane & 15;

  __shared__ _Float16 Qlds[16 * 256];     // Q^T tile  [i][c]
  __shared__ _Float16 Plds[8][16 * 32];   // per-wave P tile [i][j]
  __shared__ float    Oacc[256 * 16];     // merged O  [c][i]
  __shared__ float    mrow[8][16], lrow[8][16];
  __shared__ float    gm[16], gl[16];

  const _Float16* qb = qT + (size_t)b * CN * CC;
  const _Float16* kb = kT + (size_t)b * CN * CC;
  const _Float16* vb = vN + (size_t)b * CC * CN;

  {  // stage Q^T tile, zero merge buffer, init per-wave softmax state
    int i = t >> 4, c0 = (t & 15) * 16;
    const _Float16* p = qb + (size_t)(i0 + i) * CC + c0;
    *(v8h*)(&Qlds[i * 256 + c0])     = *(const v8h*)p;
    *(v8h*)(&Qlds[i * 256 + c0 + 8]) = *(const v8h*)(p + 8);
  }
#pragma unroll
  for (int e = 0; e < 16; ++e) Oacc[t * 16 + e] = 0.f;
  if (lane < 16) { mrow[wid][lane] = -1e30f; lrow[wid][lane] = 0.f; }
  __syncthreads();

  v8f acc[16];
#pragma unroll
  for (int ct = 0; ct < 16; ++ct) acc[ct] = vzero();

  for (int it = 0; it < 4; ++it) {
    int j0 = (wid + it * 8) * 32;
    if (it < 3) {  // prefetch next K/V tiles into caches
      int jn = (wid + (it + 1) * 8) * 32;
      __builtin_prefetch(kb + (size_t)(jn + lcol) * CC, 0, 1);
      __builtin_prefetch(vb + (size_t)(lane * 8) * CN + jn, 0, 1);
    }

    // ---- S tiles: [16 i x 32 j], K-loop over 256 channels ----
    v8f S0 = vzero(), S1 = vzero();
#pragma unroll
    for (int kc = 0; kc < CC; kc += 32) {
      const _Float16* pa = &Qlds[lcol * 256 + kc + 16 * hl];
      v16h Aq = ld16(pa, pa + 8);
      const _Float16* p0 = kb + (size_t)(j0 + lcol) * CC + kc + 16 * hl;
      const _Float16* p1 = kb + (size_t)(j0 + 16 + lcol) * CC + kc + 16 * hl;
      v16h B0 = ld16(p0, p0 + 8);
      v16h B1 = ld16(p1, p1 + 8);
      S0 = wmma16(Aq, B0, S0);
      S1 = wmma16(Aq, B1, S1);
    }

    // ---- online softmax (rows m = 8*hl + r live across the 16-lane half) ----
    asm volatile("" ::: "memory");
    float oldm_lane = mrow[wid][lcol];
    float oldm[8], newm[8], P0[8], P1[8];
#pragma unroll
    for (int r = 0; r < 8; ++r) {
      float tm = red16_max(fmaxf(S0[r], S1[r]));
      oldm[r] = mrow[wid][8 * hl + r];
      newm[r] = fmaxf(oldm[r], tm);
    }
#pragma unroll
    for (int r = 0; r < 8; ++r) {
      P0[r] = __expf(S0[r] - newm[r]);
      P1[r] = __expf(S1[r] - newm[r]);
      float rsum = red16_sum(P0[r] + P1[r]);
      float newl = lrow[wid][8 * hl + r] * __expf(oldm[r] - newm[r]) + rsum;
      mrow[wid][8 * hl + r] = newm[r];
      lrow[wid][8 * hl + r] = newl;
      Plds[wid][(8 * hl + r) * 32 + lcol]      = (_Float16)P0[r];
      Plds[wid][(8 * hl + r) * 32 + 16 + lcol] = (_Float16)P1[r];
    }
    asm volatile("" ::: "memory");
    float newm_lane = mrow[wid][lcol];
    float fi = __expf(oldm_lane - newm_lane);  // rescale for O column i = lcol

    // ---- O += V_j * P^T  (K = 32 j's), 16 c-tiles per wave ----
    const _Float16* pb = &Plds[wid][lcol * 32 + 16 * hl];
    v16h Bp = ld16(pb, pb + 8);
#pragma unroll
    for (int ct = 0; ct < 16; ++ct) {
      const _Float16* vp = vb + (size_t)(ct * 16 + lcol) * CN + j0 + 16 * hl;
      v16h Av = ld16(vp, vp + 8);
#pragma unroll
      for (int e = 0; e < 8; ++e) acc[ct][e] *= fi;
      acc[ct] = wmma16(Av, Bp, acc[ct]);
    }
  }

  // ---- merge the 8 per-wave partials ----
  __syncthreads();
  if (t < 16) {
    float m = -1e30f;
#pragma unroll
    for (int w = 0; w < 8; ++w) m = fmaxf(m, mrow[w][t]);
    float l = 0.f;
#pragma unroll
    for (int w = 0; w < 8; ++w) l += lrow[w][t] * __expf(mrow[w][t] - m);
    gm[t] = m;
    gl[t] = l;
  }
  __syncthreads();
  float fw = __expf(mrow[wid][lcol] - gm[lcol]);
#pragma unroll
  for (int ct = 0; ct < 16; ++ct)
#pragma unroll
    for (int r = 0; r < 8; ++r)
      atomicAdd(&Oacc[(ct * 16 + 8 * hl + r) * 16 + lcol], acc[ct][r] * fw);
  __syncthreads();

  {  // normalize and store transposed [i][c] f16 (feeds output projection)
    int i = t >> 4, c0 = (t & 15) * 16;
    float inv = 1.f / gl[i];
    v8h o0, o1;
#pragma unroll
    for (int e = 0; e < 8; ++e) o0[e] = (_Float16)(Oacc[(c0 + e) * 16 + i] * inv);
#pragma unroll
    for (int e = 0; e < 8; ++e) o1[e] = (_Float16)(Oacc[(c0 + 8 + e) * 16 + i] * inv);
    _Float16* dst = ovT + (size_t)b * CN * CC + (size_t)(i0 + i) * CC + c0;
    *(v8h*)dst = o0;
    *(v8h*)(dst + 8) = o1;
  }
}

// ---------------------------------------------------------------------------
extern "C" void kernel_launch(void* const* d_in, const int* in_sizes, int n_in,
                              void* d_out, int out_size, void* d_ws, size_t ws_size,
                              hipStream_t stream) {
  (void)in_sizes; (void)n_in; (void)out_size; (void)ws_size;
  const float* x     = (const float*)d_in[0];
  const float* gamma = (const float*)d_in[1];
  const float* beta  = (const float*)d_in[2];
  const float* wq    = (const float*)d_in[3];
  const float* bq    = (const float*)d_in[4];
  const float* wk    = (const float*)d_in[5];
  const float* bk    = (const float*)d_in[6];
  const float* wv    = (const float*)d_in[7];
  const float* bv    = (const float*)d_in[8];
  const float* wp    = (const float*)d_in[9];
  const float* bp    = (const float*)d_in[10];
  float* out = (float*)d_out;

  char* ws = (char*)d_ws;
  const size_t W16SZ = (size_t)CC * CC * 2;       // 128 KB per matrix
  const size_t TSZ   = (size_t)CB * CN * CC * 2;  // 16.78 MB per tensor
  _Float16* w16 = (_Float16*)ws;
  _Float16* hT  = (_Float16*)(ws + 4 * W16SZ);
  _Float16* qT  = (_Float16*)(ws + 4 * W16SZ + 1 * TSZ);
  _Float16* kT  = (_Float16*)(ws + 4 * W16SZ + 2 * TSZ);
  _Float16* vn  = (_Float16*)(ws + 4 * W16SZ + 3 * TSZ);
  _Float16* ovT = (_Float16*)(ws + 4 * W16SZ + 4 * TSZ);

  cvt_kernel<<<dim3(CC * CC / 256), 256, 0, stream>>>(wq, wk, wv, wp, w16);
  gn_kernel<<<dim3(32, CB), 256, 0, stream>>>(x, gamma, beta, hT);
  const float scale = 0.0625f;  // C^-0.5, folded into q
  mm_kernel<0><<<dim3(CN / 64, CB), 256, 0, stream>>>(w16,              bq, hT,  qT, nullptr, nullptr, scale);
  mm_kernel<0><<<dim3(CN / 64, CB), 256, 0, stream>>>(w16 + CC * CC,    bk, hT,  kT, nullptr, nullptr, 1.0f);
  mm_kernel<1><<<dim3(CN / 64, CB), 256, 0, stream>>>(w16 + 2 * CC * CC, bv, hT, vn, nullptr, nullptr, 1.0f);
  fa_kernel<<<dim3(CN / 16, CB), 256, 0, stream>>>(qT, kT, vn, ovT);
  mm_kernel<2><<<dim3(CN / 64, CB), 256, 0, stream>>>(w16 + 3 * CC * CC, bp, ovT, nullptr, out, x, 1.0f);
}
